// model_10359461118152
// MI455X (gfx1250) — matrix operand order repaired and checked by reading, not compiled, not run on hardware
//
#include <hip/hip_runtime.h>
#include <stdint.h>

#define N_NODES 16384
#define N_EDGES 131072
#define F_IN 6
#define S_DIM 4
#define G_GRAPHS 256
#define H_DIM 30
#define C1 32
#define C2 64
#define C3 32

typedef __bf16 v16bf __attribute__((ext_vector_type(16)));
typedef __bf16 bf2   __attribute__((ext_vector_type(2)));
typedef float  v8f   __attribute__((ext_vector_type(8)));

union Frag {
    uint4 q[2];
    v16bf v;
};
union BU {
    bf2 b;
    unsigned int u;
};

__device__ __forceinline__ unsigned short f2bf(float f) {
    unsigned int u = __float_as_uint(f);
    unsigned int r = u + 0x7FFFu + ((u >> 16) & 1u);
    return (unsigned short)(r >> 16);
}

// wave-internal LDS ordering: LDS pipe is in-order per wave (DScnt); we only
// need to stop the compiler from reordering and emit the required waits.
__device__ __forceinline__ void wave_sync() {
    __builtin_amdgcn_fence(__ATOMIC_ACQ_REL, "wavefront");
    __builtin_amdgcn_wave_barrier();
}

// ---------------------------------------------------------------- zero
__global__ void zero_kernel(float* __restrict__ p, int n) {
    int i = blockIdx.x * blockDim.x + threadIdx.x;
    if (i < n) p[i] = 0.f;
}

// ---------------------------------------------------------------- B packing (fragment-transposed bf16 layouts)
// Bg1: [6 ksteps][2 ntiles][16 n][32 k]  for ECC1 (K order q = h*6+f, then 6 bias rows, pad to 192)
__global__ void pack_b1(const float* __restrict__ wk1, const float* __restrict__ bk1,
                        unsigned short* __restrict__ Bg1) {
    int i = blockIdx.x * blockDim.x + threadIdx.x;
    if (i >= 6 * 2 * 16 * 32) return;
    int k  = i & 31;
    int n  = (i >> 5) & 15;
    int nt = (i >> 9) & 1;
    int kk = i >> 10;
    int q  = kk * 32 + k;
    int c1 = nt * 16 + n;
    float v = 0.f;
    if (q < H_DIM * F_IN) {
        int h = q / F_IN, f = q % F_IN;
        v = wk1[h * (F_IN * C1) + f * C1 + c1];
    } else if (q < H_DIM * F_IN + F_IN) {
        v = bk1[(q - H_DIM * F_IN) * C1 + c1];
    }
    Bg1[i] = f2bf(v);
}

// Bg2: [31 ksteps][4 ntiles][16 n][32 k]  for ECC2 (K order q = h*32+c1, kstep 30 = bias rows)
__global__ void pack_b2(const float* __restrict__ wk2, const float* __restrict__ bk2,
                        unsigned short* __restrict__ Bg2) {
    int i = blockIdx.x * blockDim.x + threadIdx.x;
    if (i >= 31 * 4 * 16 * 32) return;
    int k  = i & 31;
    int n  = (i >> 5) & 15;
    int nt = (i >> 9) & 3;
    int kk = i >> 11;
    int c2 = nt * 16 + n;
    float v = (kk < 30) ? wk2[kk * (C1 * C2) + k * C2 + c2]
                        : bk2[k * C2 + c2];
    Bg2[i] = f2bf(v);
}

// ---------------------------------------------------------------- edge filter-net MLP (both layers)
__device__ __forceinline__ void run_net(const float ef[S_DIM],
                                        const float* __restrict__ w0, const float* __restrict__ b0,
                                        const float* __restrict__ w1, const float* __restrict__ b1,
                                        float* __restrict__ out) {
    float h0[H_DIM];
#pragma unroll
    for (int j = 0; j < H_DIM; ++j) {
        float a = b0[j];
#pragma unroll
        for (int s = 0; s < S_DIM; ++s) a = fmaf(ef[s], w0[s * H_DIM + j], a);
        h0[j] = fmaxf(a, 0.f);
    }
#pragma unroll 5
    for (int j = 0; j < H_DIM; ++j) {
        float a = b1[j];
#pragma unroll
        for (int k = 0; k < H_DIM; ++k) a = fmaf(h0[k], w1[k * H_DIM + j], a);
        out[j] = fmaxf(a, 0.f);
    }
}

__global__ void edge_mlp(const float* __restrict__ efeat,
                         const float* __restrict__ w0a, const float* __restrict__ b0a,
                         const float* __restrict__ w1a, const float* __restrict__ b1a,
                         const float* __restrict__ w0b, const float* __restrict__ b0b,
                         const float* __restrict__ w1b, const float* __restrict__ b1b,
                         float* __restrict__ h1out, float* __restrict__ h2out) {
    int e = blockIdx.x * blockDim.x + threadIdx.x;
    if (e >= N_EDGES) return;
    float ef[S_DIM];
#pragma unroll
    for (int s = 0; s < S_DIM; ++s) ef[s] = efeat[(size_t)e * S_DIM + s];
    run_net(ef, w0a, b0a, w1a, b1a, h1out + (size_t)e * H_DIM);
    run_net(ef, w0b, b0b, w1b, b1b, h2out + (size_t)e * H_DIM);
}

// ---------------------------------------------------------------- ECC1 fused GEMM + scatter (WMMA bf16)
// packed A pair for compile-time even K index q (pairs never straddle region
// boundaries: 180 and 186 are even). Operands already bf16 in registers.
__device__ __forceinline__ unsigned int a1pair(int q, const __bf16* xb, const __bf16* hb) {
    BU t;
    if (q < H_DIM * F_IN) {
        bf2 h, x;
        h[0] = hb[q / F_IN];       h[1] = hb[(q + 1) / F_IN];
        x[0] = xb[q % F_IN];       x[1] = xb[(q + 1) % F_IN];
        t.b = h * x;                       // packed bf16 multiply
    } else if (q < H_DIM * F_IN + F_IN) {
        t.b[0] = xb[q - H_DIM * F_IN];
        t.b[1] = xb[q + 1 - H_DIM * F_IN];
    } else {
        t.u = 0;
    }
    return t.u;
}

__global__ __launch_bounds__(256) void ecc1_kernel(
        const float* __restrict__ x, const int* __restrict__ src, const int* __restrict__ tgt,
        const float* __restrict__ h1, const unsigned short* __restrict__ Bg1,
        float* __restrict__ agg1) {
    __shared__ __align__(16) unsigned int Atile[8][16][16];
    __shared__ int tglds[8][16];

    const int lane = threadIdx.x & 31;
    const int wv   = threadIdx.x >> 5;
    const int e0   = (blockIdx.x * 8 + wv) * 16;
    const int laneHi = lane >> 4;
    const int laneLo = lane & 15;

    if (lane < 16) tglds[wv][lane] = tgt[e0 + lane];

    // register-resident bf16 rows for this lane's A-row (converted once, RNE via fptrunc)
    const int erow = e0 + laneLo;
    __bf16 xb[F_IN], hb[H_DIM];
    {
        int s = src[erow];
#pragma unroll
        for (int f = 0; f < F_IN; ++f) xb[f] = (__bf16)x[(size_t)s * F_IN + f];
#pragma unroll
        for (int j = 0; j < H_DIM; ++j) hb[j] = (__bf16)h1[(size_t)erow * H_DIM + j];
    }

    v8f acc[2] = {};

#pragma unroll
    for (int kk = 0; kk < 6; ++kk) {
        wave_sync();
        {   // build 8 packed bf16 pairs; single cndmask per pair for the laneHi variant
            unsigned int u[8];
#pragma unroll
            for (int i = 0; i < 8; ++i) {
                const int q = kk * 32 + 2 * i;
                unsigned int lo = a1pair(q, xb, hb);
                unsigned int hi = a1pair(q + 16, xb, hb);
                u[i] = laneHi ? hi : lo;
            }
            uint4* arow = (uint4*)&Atile[wv][laneLo][0];
            arow[laneHi * 2]     = make_uint4(u[0], u[1], u[2], u[3]);
            arow[laneHi * 2 + 1] = make_uint4(u[4], u[5], u[6], u[7]);
        }
        wave_sync();
        Frag a;
        {
            const uint4* arow = (const uint4*)&Atile[wv][laneLo][0];
            a.q[0] = arow[laneHi * 2];
            a.q[1] = arow[laneHi * 2 + 1];
        }
#pragma unroll
        for (int nt = 0; nt < 2; ++nt) {
            Frag b;
            const uint4* bp = (const uint4*)(Bg1 + (((size_t)(kk * 2 + nt) * 16 + laneLo) * 32 + laneHi * 16));
            b.q[0] = bp[0];
            b.q[1] = bp[1];
            acc[nt] = __builtin_amdgcn_wmma_f32_16x16x32_bf16(
                false, a.v, false, b.v, (short)0, acc[nt], false, false);
        }
    }

    wave_sync();
    int col = laneLo;
#pragma unroll
    for (int j = 0; j < 8; ++j) {
        int m  = j + laneHi * 8;
        int tg = tglds[wv][m];
        atomicAdd(&agg1[(size_t)tg * C1 + col],      acc[0][j]);
        atomicAdd(&agg1[(size_t)tg * C1 + 16 + col], acc[1][j]);
    }
}

// ---------------------------------------------------------------- ECC2 fused GEMM + scatter (WMMA bf16, K=992)
__global__ __launch_bounds__(256) void ecc2_kernel(
        const float* __restrict__ x1, const int* __restrict__ src, const int* __restrict__ tgt,
        const float* __restrict__ h2, const unsigned short* __restrict__ Bg2,
        float* __restrict__ agg2) {
    __shared__ __align__(16) unsigned int Atile[8][16][16];
    __shared__ float hlds[8][16][H_DIM];
    __shared__ int   tglds[8][16];

    const int lane = threadIdx.x & 31;
    const int wv   = threadIdx.x >> 5;
    const int e0   = (blockIdx.x * 8 + wv) * 16;
    const int laneHi = lane >> 4;
    const int laneLo = lane & 15;
    const int cb = laneHi * 16;

    if (lane < 16) tglds[wv][lane] = tgt[e0 + lane];

    // register-resident: this lane's 16 x1 columns, pre-packed as 8 bf16x2 (RNE once)
    BU x1p[8];
    {
        int s = src[e0 + laneLo];
        const float4* xr = (const float4*)(x1 + (size_t)s * C1 + cb);
#pragma unroll
        for (int i = 0; i < 4; ++i) {
            float4 v = xr[i];
            x1p[2 * i].b[0]     = (__bf16)v.x;
            x1p[2 * i].b[1]     = (__bf16)v.y;
            x1p[2 * i + 1].b[0] = (__bf16)v.z;
            x1p[2 * i + 1].b[1] = (__bf16)v.w;
        }
    }
    // h row needs runtime kk indexing -> stage per-wave in LDS
    for (int t = lane; t < 16 * H_DIM; t += 32) {
        int r = t / H_DIM, j = t % H_DIM;
        hlds[wv][r][j] = h2[(size_t)(e0 + r) * H_DIM + j];
    }

    v8f acc[4] = {};

    for (int kk = 0; kk < 31; ++kk) {
        if (kk + 1 < 31)  // pull next k-step's B slice toward the WGP
            __builtin_prefetch(Bg2 + (size_t)(kk + 1) * 4 * 16 * 32, 0, 1);
        wave_sync();
        {   // A tile row: h2[row][kk] (bf16 broadcast) * packed x1 pairs; k-step 30 = bias rows
            unsigned int u[8];
            if (kk < 30) {
                __bf16 hbv = (__bf16)hlds[wv][laneLo][kk];
                bf2 hh;
                hh[0] = hbv;
                hh[1] = hbv;
#pragma unroll
                for (int i = 0; i < 8; ++i) {
                    BU t;
                    t.b = hh * x1p[i].b;       // packed bf16 multiply
                    u[i] = t.u;
                }
            } else {
#pragma unroll
                for (int i = 0; i < 8; ++i) u[i] = x1p[i].u;
            }
            uint4* arow = (uint4*)&Atile[wv][laneLo][0];
            arow[laneHi * 2]     = make_uint4(u[0], u[1], u[2], u[3]);
            arow[laneHi * 2 + 1] = make_uint4(u[4], u[5], u[6], u[7]);
        }
        wave_sync();
        Frag a;
        {
            const uint4* arow = (const uint4*)&Atile[wv][laneLo][0];
            a.q[0] = arow[laneHi * 2];
            a.q[1] = arow[laneHi * 2 + 1];
        }
#pragma unroll
        for (int nt = 0; nt < 4; ++nt) {
            Frag b;
            const uint4* bp = (const uint4*)(Bg2 + (((size_t)(kk * 4 + nt) * 16 + laneLo) * 32 + laneHi * 16));
            b.q[0] = bp[0];
            b.q[1] = bp[1];
            acc[nt] = __builtin_amdgcn_wmma_f32_16x16x32_bf16(
                false, a.v, false, b.v, (short)0, acc[nt], false, false);
        }
    }

    wave_sync();
    int col = laneLo;
#pragma unroll
    for (int j = 0; j < 8; ++j) {
        int m  = j + laneHi * 8;
        int tg = tglds[wv][m];
#pragma unroll
        for (int nt = 0; nt < 4; ++nt)
            atomicAdd(&agg2[(size_t)tg * C2 + nt * 16 + col], acc[nt][j]);
    }
}

// ---------------------------------------------------------------- node updates
__global__ void node_x1(const float* __restrict__ x, const float* __restrict__ agg1,
                        const float* __restrict__ root, const float* __restrict__ bias,
                        float* __restrict__ x1) {
    int i = blockIdx.x * blockDim.x + threadIdx.x;
    if (i >= N_NODES * C1) return;
    int n = i >> 5, c = i & 31;
    float a = agg1[i] + bias[c];
#pragma unroll
    for (int f = 0; f < F_IN; ++f) a = fmaf(x[n * F_IN + f], root[f * C1 + c], a);
    x1[i] = fmaxf(a, 0.f);
}

__global__ void node_x2(const float* __restrict__ x1, const float* __restrict__ agg2,
                        const float* __restrict__ root, const float* __restrict__ bias,
                        float* __restrict__ x2) {
    int i = blockIdx.x * blockDim.x + threadIdx.x;
    if (i >= N_NODES * C2) return;
    int n = i >> 6, c = i & 63;
    float a = agg2[i] + bias[c];
#pragma unroll
    for (int k = 0; k < C1; ++k) a = fmaf(x1[n * C1 + k], root[k * C2 + c], a);
    x2[i] = fmaxf(a, 0.f);
}

__global__ void node_xw(const float* __restrict__ x2, const float* __restrict__ W,
                        float* __restrict__ xw) {
    int i = blockIdx.x * blockDim.x + threadIdx.x;
    if (i >= N_NODES * C3) return;
    int n = i >> 5, c = i & 31;
    float a = 0.f;
#pragma unroll
    for (int k = 0; k < C2; ++k) a = fmaf(x2[n * C2 + k], W[k * C3 + c], a);
    xw[i] = a;
}

__global__ void gcn_scatter(const float* __restrict__ xw, const int* __restrict__ gsrc,
                            const int* __restrict__ gtgt, const float* __restrict__ gw,
                            float* __restrict__ agg3) {
    int i = blockIdx.x * blockDim.x + threadIdx.x;
    if (i >= (N_EDGES + N_NODES) * C3) return;
    int e = i >> 5, c = i & 31;
    float v = gw[e] * xw[(size_t)gsrc[e] * C3 + c];
    atomicAdd(&agg3[(size_t)gtgt[e] * C3 + c], v);
}

__global__ void finalize_pool(const float* __restrict__ agg3, const float* __restrict__ gb,
                              const int* __restrict__ seg, float* __restrict__ out) {
    int i = blockIdx.x * blockDim.x + threadIdx.x;
    if (i >= N_NODES * C3) return;
    int n = i >> 5, c = i & 31;
    float v = fmaxf(agg3[i] + gb[c], 0.f);
    atomicAdd(&out[(size_t)seg[n] * C3 + c], v);
}

// ---------------------------------------------------------------- launch
extern "C" void kernel_launch(void* const* d_in, const int* in_sizes, int n_in,
                              void* d_out, int out_size, void* d_ws, size_t ws_size,
                              hipStream_t stream) {
    (void)in_sizes; (void)n_in; (void)out_size; (void)ws_size;

    const float* x       = (const float*)d_in[0];
    const float* e       = (const float*)d_in[1];
    const int*   src     = (const int*)d_in[2];
    const int*   tgt     = (const int*)d_in[3];
    const int*   seg     = (const int*)d_in[4];
    const int*   gsrc    = (const int*)d_in[5];
    const int*   gtgt    = (const int*)d_in[6];
    const float* gw      = (const float*)d_in[7];
    const float* e1_w0   = (const float*)d_in[8];
    const float* e1_b0   = (const float*)d_in[9];
    const float* e1_w1   = (const float*)d_in[10];
    const float* e1_b1   = (const float*)d_in[11];
    const float* e1_wk   = (const float*)d_in[12];
    const float* e1_bk   = (const float*)d_in[13];
    const float* e1_root = (const float*)d_in[14];
    const float* e1_bias = (const float*)d_in[15];
    const float* e2_w0   = (const float*)d_in[16];
    const float* e2_b0   = (const float*)d_in[17];
    const float* e2_w1   = (const float*)d_in[18];
    const float* e2_b1   = (const float*)d_in[19];
    const float* e2_wk   = (const float*)d_in[20];
    const float* e2_bk   = (const float*)d_in[21];
    const float* e2_root = (const float*)d_in[22];
    const float* e2_bias = (const float*)d_in[23];
    const float* gcn_W   = (const float*)d_in[24];
    const float* gcn_b   = (const float*)d_in[25];
    float* out = (float*)d_out;

    char* ws = (char*)d_ws;
    size_t off = 0;
    auto alloc = [&](size_t bytes) -> void* {
        void* p = ws + off;
        off += (bytes + 255) & ~(size_t)255;
        return p;
    };
    // agg1/agg2/agg3 contiguous (each region is 256B-multiple) so one zero pass covers all
    float* agg1 = (float*)alloc((size_t)N_NODES * C1 * 4);
    float* agg2 = (float*)alloc((size_t)N_NODES * C2 * 4);
    float* agg3 = (float*)alloc((size_t)N_NODES * C3 * 4);
    float* x1   = (float*)alloc((size_t)N_NODES * C1 * 4);
    float* x2   = (float*)alloc((size_t)N_NODES * C2 * 4);
    float* xw   = (float*)alloc((size_t)N_NODES * C3 * 4);
    float* h1   = (float*)alloc((size_t)N_EDGES * H_DIM * 4);
    float* h2   = (float*)alloc((size_t)N_EDGES * H_DIM * 4);
    unsigned short* Bg1 = (unsigned short*)alloc((size_t)6 * 2 * 16 * 32 * 2);
    unsigned short* Bg2 = (unsigned short*)alloc((size_t)31 * 4 * 16 * 32 * 2);

    const int T = 256;
    const int aggTotal = N_NODES * (C1 + C2 + C3);

    zero_kernel<<<aggTotal / T, T, 0, stream>>>(agg1, aggTotal);
    zero_kernel<<<(G_GRAPHS * C3) / T, T, 0, stream>>>(out, G_GRAPHS * C3);

    pack_b1<<<(6 * 2 * 16 * 32) / T, T, 0, stream>>>(e1_wk, e1_bk, Bg1);
    pack_b2<<<(31 * 4 * 16 * 32) / T, T, 0, stream>>>(e2_wk, e2_bk, Bg2);

    edge_mlp<<<N_EDGES / T, T, 0, stream>>>(e, e1_w0, e1_b0, e1_w1, e1_b1,
                                            e2_w0, e2_b0, e2_w1, e2_b1, h1, h2);

    ecc1_kernel<<<N_EDGES / (16 * 8), T, 0, stream>>>(x, src, tgt, h1, Bg1, agg1);
    node_x1<<<(N_NODES * C1) / T, T, 0, stream>>>(x, agg1, e1_root, e1_bias, x1);

    ecc2_kernel<<<N_EDGES / (16 * 8), T, 0, stream>>>(x1, src, tgt, h2, Bg2, agg2);
    node_x2<<<(N_NODES * C2) / T, T, 0, stream>>>(x1, agg2, e2_root, e2_bias, x2);

    node_xw<<<(N_NODES * C3) / T, T, 0, stream>>>(x2, gcn_W, xw);
    gcn_scatter<<<((N_EDGES + N_NODES) * C3) / T, T, 0, stream>>>(xw, gsrc, gtgt, gw, agg3);
    finalize_pool<<<(N_NODES * C3) / T, T, 0, stream>>>(agg3, gcn_b, seg, out);
}
